// AbsSeq2SeqLSTM_90039694393901
// MI455X (gfx1250) — matrix operand order repaired and checked by reading, not compiled, not run on hardware
//
#include <hip/hip_runtime.h>
#include <math.h>

// Problem constants (match reference)
#define B_    256
#define S_IN  128
#define T_OUT 128
#define H_    512
#define OUTV  1024

typedef __bf16 bf16_t;
typedef bf16_t v8bf  __attribute__((ext_vector_type(8)));
typedef bf16_t v16bf __attribute__((ext_vector_type(16)));
typedef float  v8f   __attribute__((ext_vector_type(8)));

union FragU { v16bf v; v8bf h[2]; };

__device__ __forceinline__ bf16_t f2bf(float f) {
  union { float f; unsigned u; } x; x.f = f;
  unsigned r = (x.u + 0x7FFFu + ((x.u >> 16) & 1u)) >> 16;
  union { unsigned short s; bf16_t b; } y; y.s = (unsigned short)r;
  return y.b;
}
__device__ __forceinline__ float bf2f(bf16_t b) {
  union { unsigned short s; bf16_t b; } x; x.b = b;
  union { unsigned u; float f; } y; y.u = ((unsigned)x.s) << 16;
  return y.f;
}

// A fragment (16x32 bf16): lane l holds row m0+(l&15); 16 contiguous K values [kb..kb+15],
// kb = k0 + (l>>4)*16 (matches B fragment K order).
__device__ __forceinline__ v16bf load_a_frag(const bf16_t* A, long lda, int row, int kb) {
  FragU f;
  const bf16_t* p = A + (long)row * lda + kb;
  f.h[0] = *(const v8bf*)(p);
  f.h[1] = *(const v8bf*)(p + 8);
  return f.v;
}
// B fragment (32x16) from N-major (N x K): lane l holds col n0+(l&15),
// 16 contiguous K values at k0 + (l>>4)*16 (same K order as the A fragment).
__device__ __forceinline__ v16bf load_b_frag(const bf16_t* Bt, long ldb, int col, int ks) {
  FragU f;
  const bf16_t* p = Bt + (long)col * ldb + ks;
  f.h[0] = *(const v8bf*)(p);
  f.h[1] = *(const v8bf*)(p + 8);
  return f.v;
}
#define WMMA_BF16(a, b, c) \
  __builtin_amdgcn_wmma_f32_16x16x32_bf16(false, (a), false, (b), (short)0, (c), false, false)

// ---------------- Big parallel precompute GEMM: Cbf = A @ Bt^T (bf16 out) ----------------
// Block 256 thr = 8 waves as 2x4; block tile 32x256; wave tile 16x64. grid(N/256, M/32).
__global__ __launch_bounds__(256)
void gemm_precompute(const bf16_t* __restrict__ A, long lda, int K,
                     const bf16_t* __restrict__ Bt, long ldb,
                     bf16_t* __restrict__ Cbf, int N) {
  const int lane = threadIdx.x & 31;
  const int wave = threadIdx.x >> 5;
  const int m0 = blockIdx.y * 32 + (wave >> 2) * 16;
  const int n0 = blockIdx.x * 256 + (wave & 3) * 64;
  const int arow = m0 + (lane & 15);
  const int kb8  = (lane >> 4) * 8;
  const int kb16 = (lane >> 4) * 16;
  const int bcol = n0 + (lane & 15);
  v8f acc0 = {}, acc1 = {}, acc2 = {}, acc3 = {};
  for (int k0 = 0; k0 < K; k0 += 32) {
    v16bf a  = load_a_frag(A, lda, arow, k0 + kb16);
    v16bf b0 = load_b_frag(Bt, ldb, bcol +  0, k0 + kb16);
    v16bf b1 = load_b_frag(Bt, ldb, bcol + 16, k0 + kb16);
    v16bf b2 = load_b_frag(Bt, ldb, bcol + 32, k0 + kb16);
    v16bf b3 = load_b_frag(Bt, ldb, bcol + 48, k0 + kb16);
    acc0 = WMMA_BF16(a, b0, acc0);
    acc1 = WMMA_BF16(a, b1, acc1);
    acc2 = WMMA_BF16(a, b2, acc2);
    acc3 = WMMA_BF16(a, b3, acc3);
  }
  const int rbase = m0 + (lane >> 4) * 8;
  const int cl = lane & 15;
  v8f accs[4] = {acc0, acc1, acc2, acc3};
  #pragma unroll
  for (int j = 0; j < 4; ++j) {
    const int col = n0 + j * 16 + cl;
    #pragma unroll
    for (int v = 0; v < 8; ++v)
      Cbf[(long)(rbase + v) * N + col] = f2bf(accs[j][v]);
  }
}

// ---------------- Fused LSTM step: z-GEMM + gates + state update ----------------
// Weights/bias/zx are in gate-PERMUTED column space: n' = 4*u + gate (gate: 0=i,1=f,2=g,3=o).
// C = [A0 | A1] @ Bt^T + biasP (+ ZaddP). Block tile 16x256 (wave tile 16x32); grid(8, 16).
// Each block owns 64 complete hidden units x 16 rows -> full gate fusion in-block.
// NOTE: hbf_out must not alias A0/A1 (double-buffered by caller); h/c written once per elem.
__global__ __launch_bounds__(256)
void lstm_step_gemm(const bf16_t* __restrict__ A0, long lda0, int K0,
                    const bf16_t* __restrict__ A1, long lda1, int K1,
                    const bf16_t* __restrict__ Bt, long ldb,
                    const float* __restrict__ biasP,
                    const bf16_t* __restrict__ ZaddP, long ldz,
                    float* __restrict__ h, float* __restrict__ c,
                    bf16_t* __restrict__ hbf_out) {
  __shared__ float zt[8][16][33];   // [wave][row][col], padded vs bank conflicts
  const int lane = threadIdx.x & 31;
  const int wave = threadIdx.x >> 5;
  const int m0 = blockIdx.y * 16;
  const int n0 = blockIdx.x * 256 + wave * 32;
  const int arow = m0 + (lane & 15);
  const int kb8  = (lane >> 4) * 8;
  const int kb16 = (lane >> 4) * 16;
  const int bcol = n0 + (lane & 15);
  v8f acc0 = {}, acc1 = {};
  const int K = K0 + K1;
  for (int k0 = 0; k0 < K; k0 += 32) {
    v16bf a = (k0 < K0) ? load_a_frag(A0, lda0, arow, k0 + kb16)
                        : load_a_frag(A1, lda1, arow, (k0 - K0) + kb16);
    v16bf b0 = load_b_frag(Bt, ldb, bcol,      k0 + kb16);
    v16bf b1 = load_b_frag(Bt, ldb, bcol + 16, k0 + kb16);
    acc0 = WMMA_BF16(a, b0, acc0);
    acc1 = WMMA_BF16(a, b1, acc1);
  }
  const int rbase = (lane >> 4) * 8;
  const int cl = lane & 15;
  v8f accs[2] = {acc0, acc1};
  #pragma unroll
  for (int j = 0; j < 2; ++j) {
    const int colG = n0 + j * 16 + cl;   // global permuted column
    const float bv = biasP[colG];
    #pragma unroll
    for (int v = 0; v < 8; ++v) {
      float val = accs[j][v] + bv;
      if (ZaddP) val += bf2f(ZaddP[(long)(m0 + rbase + v) * ldz + colG]);
      zt[wave][rbase + v][j * 16 + cl] = val;
    }
  }
  __syncthreads();
  // Gate fusion: 1024 (row,unit) pairs in this block; 4 per thread.
  #pragma unroll
  for (int p = 0; p < 4; ++p) {
    const int idx = p * 256 + threadIdx.x;
    const int u = idx & 63;        // local unit
    const int row = idx >> 6;
    const int w = u >> 3, base = (u & 7) * 4;
    const float zi = zt[w][row][base + 0];
    const float zf = zt[w][row][base + 1];
    const float zg = zt[w][row][base + 2];
    const float zo = zt[w][row][base + 3];
    const float ig = 1.0f / (1.0f + expf(-zi));
    const float fg = 1.0f / (1.0f + expf(-zf));
    const float g  = tanhf(zg);
    const float og = 1.0f / (1.0f + expf(-zo));
    const long ci = (long)(m0 + row) * H_ + (blockIdx.x * 64 + u);
    const float cn = fg * c[ci] + ig * g;
    const float hn = og * tanhf(cn);
    c[ci] = cn;
    h[ci] = hn;
    hbf_out[ci] = f2bf(hn);
  }
}

// ---------------- Fused logits GEMM + softmax ----------------
// Block = 16 rows x all 1024 cols, K=512; 8 waves each own 128 cols (8 WMMA accums).
// Dynamic LDS: 16*1024 logits + 256 reduction scratch floats.
__global__ __launch_bounds__(256)
void logits_softmax(const bf16_t* __restrict__ hbf, const bf16_t* __restrict__ Wout_t,
                    const float* __restrict__ b_out, float* __restrict__ out, int t) {
  extern __shared__ float smem[];
  float* S = smem;                 // [16][1024]
  float* red = smem + 16 * 1024;   // [256]
  const int lane = threadIdx.x & 31;
  const int wave = threadIdx.x >> 5;
  const int m0 = blockIdx.x * 16;
  const int n0 = wave * 128;
  const int arow = m0 + (lane & 15);
  const int kb8  = (lane >> 4) * 8;
  const int kb16 = (lane >> 4) * 16;
  const int bcol = n0 + (lane & 15);
  v8f acc[8];
  #pragma unroll
  for (int j = 0; j < 8; ++j) acc[j] = (v8f){};
  for (int k0 = 0; k0 < H_; k0 += 32) {
    v16bf a = load_a_frag(hbf, H_, arow, k0 + kb16);
    #pragma unroll
    for (int j = 0; j < 8; ++j) {
      v16bf b = load_b_frag(Wout_t, H_, bcol + j * 16, k0 + kb16);
      acc[j] = WMMA_BF16(a, b, acc[j]);
    }
  }
  const int rloc = (lane >> 4) * 8;
  const int cl = lane & 15;
  #pragma unroll
  for (int j = 0; j < 8; ++j) {
    const int col = n0 + j * 16 + cl;
    const float bv = b_out[col];
    #pragma unroll
    for (int v = 0; v < 8; ++v)
      S[(rloc + v) * 1024 + col] = acc[j][v] + bv;
  }
  __syncthreads();
  const int r = threadIdx.x >> 4;
  const int q = threadIdx.x & 15;
  const float* row = S + r * 1024;
  float m = -INFINITY;
  for (int n = q; n < OUTV; n += 16) m = fmaxf(m, row[n]);
  red[threadIdx.x] = m;
  __syncthreads();
  #pragma unroll
  for (int s = 8; s > 0; s >>= 1) {
    if (q < s) red[threadIdx.x] = fmaxf(red[threadIdx.x], red[threadIdx.x + s]);
    __syncthreads();
  }
  m = red[r * 16];
  __syncthreads();
  float sum = 0.0f;
  for (int n = q; n < OUTV; n += 16) sum += expf(row[n] - m);
  red[threadIdx.x] = sum;
  __syncthreads();
  #pragma unroll
  for (int s = 8; s > 0; s >>= 1) {
    if (q < s) red[threadIdx.x] += red[threadIdx.x + s];
    __syncthreads();
  }
  const float inv = 1.0f / red[r * 16];
  float* o = out + ((long)(m0 + r) * T_OUT + t) * OUTV;
  for (int n = q; n < OUTV; n += 16) o[n] = expf(row[n] - m) * inv;
}

// ---------------- Prep kernels ----------------
__global__ void gather_rows_bf16(const int* __restrict__ idx, const float* __restrict__ table,
                                 int rowlen, bf16_t* __restrict__ out, long total) {
  for (long i = blockIdx.x * (long)blockDim.x + threadIdx.x; i < total;
       i += (long)gridDim.x * blockDim.x) {
    long r = i / rowlen; int k = (int)(i - r * rowlen);
    out[i] = f2bf(table[(long)idx[r] * rowlen + k]);
  }
}

// D (N x (K0+K1), bf16) = [S0; S1]^T, optionally with gate-permuted rows:
// permuted row np reads original column n = (np&3)*(N/4) + (np>>2).
__global__ void convT_combine_bf16(const float* __restrict__ S0, int K0,
                                   const float* __restrict__ S1, int K1,
                                   int N, bf16_t* __restrict__ D, int permute) {
  const long K = (long)K0 + K1;
  const long total = (long)N * K;
  for (long i = blockIdx.x * (long)blockDim.x + threadIdx.x; i < total;
       i += (long)gridDim.x * blockDim.x) {
    long np = i / K; long k = i - np * K;
    long n = permute ? (long)((np & 3) * (N >> 2) + (np >> 2)) : np;
    float v = (k < K0) ? S0[k * N + n] : S1[(k - K0) * N + n];
    D[i] = f2bf(v);
  }
}

__global__ void permute_bias(const float* __restrict__ b, float* __restrict__ bp, int N) {
  int i = blockIdx.x * blockDim.x + threadIdx.x;
  if (i < N) bp[i] = b[(i & 3) * (N >> 2) + (i >> 2)];
}

__global__ void fill_zero_u32(unsigned int* __restrict__ p, long n) {
  for (long i = blockIdx.x * (long)blockDim.x + threadIdx.x; i < n;
       i += (long)gridDim.x * blockDim.x) p[i] = 0u;
}

__global__ void combine_hc(const float* __restrict__ hf, const float* __restrict__ cf,
                           const float* __restrict__ hb, const float* __restrict__ cb,
                           float* __restrict__ hd, float* __restrict__ cd,
                           bf16_t* __restrict__ hdbf) {
  int idx = blockIdx.x * blockDim.x + threadIdx.x;
  if (idx >= B_ * H_) return;
  float hv = hf[idx] + hb[idx];
  float cv = cf[idx] + cb[idx];
  hd[idx] = hv; cd[idx] = cv; hdbf[idx] = f2bf(hv);
}

extern "C" void kernel_launch(void* const* d_in, const int* in_sizes, int n_in,
                              void* d_out, int out_size, void* d_ws, size_t ws_size,
                              hipStream_t stream) {
  (void)in_sizes; (void)n_in; (void)out_size;
  const int*   inputs  = (const int*)d_in[0];
  const int*   teacher = (const int*)d_in[1];
  const float* in_emb  = (const float*)d_in[3];
  const float* out_emb = (const float*)d_in[4];
  const float* Wx_f = (const float*)d_in[5];
  const float* Wh_f = (const float*)d_in[6];
  const float* b_f  = (const float*)d_in[7];
  const float* Wx_b = (const float*)d_in[8];
  const float* Wh_b = (const float*)d_in[9];
  const float* b_b  = (const float*)d_in[10];
  const float* Wx_d = (const float*)d_in[11];
  const float* Wh_d = (const float*)d_in[12];
  const float* b_d  = (const float*)d_in[13];
  const float* W_out = (const float*)d_in[14];
  const float* b_out = (const float*)d_in[15];
  float* out = (float*)d_out;

  char* ws = (char*)d_ws;
  size_t off = 0;
  auto alloc = [&](size_t bytes) -> void* {
    void* p = ws + off;
    off += (bytes + 255) & ~(size_t)255;
    return p;
  };
  // Base buffers (~116 MB)
  bf16_t* x_bf   = (bf16_t*)alloc((size_t)B_ * S_IN * H_ * 2);         // (B,S,H)
  bf16_t* y_bf   = (bf16_t*)alloc((size_t)T_OUT * B_ * OUTV * 2);      // (T,B,1024)
  bf16_t* Wf_t   = (bf16_t*)alloc((size_t)(4 * H_) * (2 * H_) * 2);    // 2048 x 1024 (perm)
  bf16_t* Wb_t   = (bf16_t*)alloc((size_t)(4 * H_) * (2 * H_) * 2);    // 2048 x 1024 (perm)
  bf16_t* Wd_t   = (bf16_t*)alloc((size_t)(4 * H_) * (OUTV + H_) * 2); // 2048 x 1536 (perm)
  bf16_t* Wout_t = (bf16_t*)alloc((size_t)OUTV * H_ * 2);              // 1024 x 512
  float*  bpf = (float*)alloc((size_t)4 * H_ * 4);
  float*  bpb = (float*)alloc((size_t)4 * H_ * 4);
  float*  bpd = (float*)alloc((size_t)4 * H_ * 4);
  float*  h_f = (float*)alloc((size_t)B_ * H_ * 4);
  float*  c_f = (float*)alloc((size_t)B_ * H_ * 4);
  float*  h_b = (float*)alloc((size_t)B_ * H_ * 4);
  float*  c_b = (float*)alloc((size_t)B_ * H_ * 4);
  float*  h_d = (float*)alloc((size_t)B_ * H_ * 4);
  float*  c_d = (float*)alloc((size_t)B_ * H_ * 4);
  bf16_t* hbfF[2] = { (bf16_t*)alloc((size_t)B_ * H_ * 2), (bf16_t*)alloc((size_t)B_ * H_ * 2) };
  bf16_t* hbfB[2] = { (bf16_t*)alloc((size_t)B_ * H_ * 2), (bf16_t*)alloc((size_t)B_ * H_ * 2) };
  bf16_t* hbfD[2] = { (bf16_t*)alloc((size_t)B_ * H_ * 2), (bf16_t*)alloc((size_t)B_ * H_ * 2) };
  // Optional big-workspace precompute buffers (bf16, gate-permuted cols): 3 x 128 MB
  bf16_t* zx_f = (bf16_t*)alloc((size_t)B_ * S_IN * 4 * H_ * 2);
  bf16_t* zx_b = (bf16_t*)alloc((size_t)B_ * S_IN * 4 * H_ * 2);
  bf16_t* zx_d = (bf16_t*)alloc((size_t)T_OUT * B_ * 4 * H_ * 2);
  const bool big = (off <= ws_size);   // deterministic path choice

  // ---- Prep ----
  gather_rows_bf16<<<4096, 256, 0, stream>>>(inputs, in_emb, H_, x_bf, (long)B_ * S_IN * H_);
  gather_rows_bf16<<<8192, 256, 0, stream>>>(teacher, out_emb, OUTV, y_bf, (long)T_OUT * B_ * OUTV);
  convT_combine_bf16<<<2048, 256, 0, stream>>>(Wx_f, H_, Wh_f, H_, 4 * H_, Wf_t, 1);
  convT_combine_bf16<<<2048, 256, 0, stream>>>(Wx_b, H_, Wh_b, H_, 4 * H_, Wb_t, 1);
  convT_combine_bf16<<<3072, 256, 0, stream>>>(Wx_d, OUTV, Wh_d, H_, 4 * H_, Wd_t, 1);
  convT_combine_bf16<<<1024, 256, 0, stream>>>(W_out, H_, nullptr, 0, OUTV, Wout_t, 0);
  permute_bias<<<8, 256, 0, stream>>>(b_f, bpf, 4 * H_);
  permute_bias<<<8, 256, 0, stream>>>(b_b, bpb, 4 * H_);
  permute_bias<<<8, 256, 0, stream>>>(b_d, bpd, 4 * H_);
  fill_zero_u32<<<512, 256, 0, stream>>>((unsigned int*)h_f, B_ * H_);
  fill_zero_u32<<<512, 256, 0, stream>>>((unsigned int*)c_f, B_ * H_);
  fill_zero_u32<<<512, 256, 0, stream>>>((unsigned int*)h_b, B_ * H_);
  fill_zero_u32<<<512, 256, 0, stream>>>((unsigned int*)c_b, B_ * H_);
  fill_zero_u32<<<256, 256, 0, stream>>>((unsigned int*)hbfF[0], B_ * H_ / 2);
  fill_zero_u32<<<256, 256, 0, stream>>>((unsigned int*)hbfB[0], B_ * H_ / 2);

  if (big) {  // hoist input projections out of the serial chain (fully parallel GEMMs)
    gemm_precompute<<<dim3(8, 1024), 256, 0, stream>>>(x_bf, (long)H_, H_,
                                                       Wf_t, (long)(2 * H_), zx_f, 4 * H_);
    gemm_precompute<<<dim3(8, 1024), 256, 0, stream>>>(x_bf, (long)H_, H_,
                                                       Wb_t, (long)(2 * H_), zx_b, 4 * H_);
    gemm_precompute<<<dim3(8, 1024), 256, 0, stream>>>(y_bf, (long)OUTV, OUTV,
                                                       Wd_t, (long)(OUTV + H_), zx_d, 4 * H_);
  }

  const dim3 sg(8, 16);  // step GEMM grid: N=2048 / 256, M=256 / 16
  const size_t sm_ls = (16 * 1024 + 256) * sizeof(float);

  // ---- Encoder: two LSTMs, 128 steps ----
  for (int t = 0; t < S_IN; ++t) {
    const int ri = t & 1, wi = (t + 1) & 1;
    if (big) {
      lstm_step_gemm<<<sg, 256, 0, stream>>>(
          hbfF[ri], (long)H_, H_, nullptr, 0, 0, Wf_t + H_, (long)(2 * H_), bpf,
          zx_f + (long)t * 4 * H_, (long)S_IN * 4 * H_, h_f, c_f, hbfF[wi]);
      lstm_step_gemm<<<sg, 256, 0, stream>>>(
          hbfB[ri], (long)H_, H_, nullptr, 0, 0, Wb_t + H_, (long)(2 * H_), bpb,
          zx_b + (long)t * 4 * H_, (long)S_IN * 4 * H_, h_b, c_b, hbfB[wi]);
    } else {
      const bf16_t* xt = x_bf + (long)t * H_;
      lstm_step_gemm<<<sg, 256, 0, stream>>>(
          xt, (long)S_IN * H_, H_, hbfF[ri], (long)H_, H_, Wf_t, (long)(2 * H_), bpf,
          nullptr, 0, h_f, c_f, hbfF[wi]);
      lstm_step_gemm<<<sg, 256, 0, stream>>>(
          xt, (long)S_IN * H_, H_, hbfB[ri], (long)H_, H_, Wb_t, (long)(2 * H_), bpb,
          nullptr, 0, h_b, c_b, hbfB[wi]);
    }
  }

  combine_hc<<<512, 256, 0, stream>>>(h_f, c_f, h_b, c_b, h_d, c_d, hbfD[0]);

  // ---- Decoder: 128 steps; step GEMM then fused logits+softmax ----
  for (int t = 0; t < T_OUT; ++t) {
    const int ri = t & 1, wi = (t + 1) & 1;
    if (big) {
      lstm_step_gemm<<<sg, 256, 0, stream>>>(
          hbfD[ri], (long)H_, H_, nullptr, 0, 0, Wd_t + OUTV, (long)(OUTV + H_), bpd,
          zx_d + (long)t * B_ * 4 * H_, (long)(4 * H_), h_d, c_d, hbfD[wi]);
    } else {
      lstm_step_gemm<<<sg, 256, 0, stream>>>(
          y_bf + (long)t * B_ * OUTV, (long)OUTV, OUTV, hbfD[ri], (long)H_, H_,
          Wd_t, (long)(OUTV + H_), bpd, nullptr, 0, h_d, c_d, hbfD[wi]);
    }
    logits_softmax<<<16, 256, sm_ls, stream>>>(hbfD[wi], Wout_t, b_out, out, t);
  }
}
